// MHSA_3453153706580
// MI455X (gfx1250) — matrix operand order repaired and checked by reading, not compiled, not run on hardware
//
#include <hip/hip_runtime.h>

// ---------------------------------------------------------------------------
// MHSA for MI455X (gfx1250, wave32, WMMA).
//   B=2, N=2048, HIDDEN=1024, H=16 heads, D=64, SCALE = 1/sqrt(1024) = 0.03125
// Pipeline:
//   cvt_hidden : fp32 -> bf16 once (was re-converted per head)
//   cvt_w      : fp32 -> bf16 + transpose to [h][e][k] once (was per row-tile)
//   qkv_proj   : bf16 WMMA GEMM, double-buffered async-to-LDS staging,
//                V written pre-transposed [b,h,d,n]
//   flash_attn : 64-key blocks, K/Vt staged with double-buffered async-to-LDS
// Workspace: Q|K|Vt (24 MB) + hbf (8 MB) + wbt (6 MB) = 38 MB.
// ---------------------------------------------------------------------------

#define B_    2
#define N_    2048
#define H_    16
#define D_    64
#define HID_  1024
#define E3_   192          // 3*D
#define SCALE_ 0.03125f
#define PER_  ((size_t)B_ * H_ * N_ * D_)   // elements per Q/K/V array (= B*N*HID)

typedef __attribute__((ext_vector_type(16))) __bf16 v16bf;
typedef __attribute__((ext_vector_type(8)))  float  v8f;

union BF16Frag { v16bf v; uint4 q[2]; };

__device__ __forceinline__ unsigned f2bf1(float x) {
  union { float f; unsigned u; } c; c.f = x;
  return (c.u + 0x7FFFu + ((c.u >> 16) & 1u)) >> 16;   // RNE
}
__device__ __forceinline__ unsigned packbf(float lo, float hi) {
  return f2bf1(lo) | (f2bf1(hi) << 16);
}

#define WMMA_BF16(a, b, c) \
  __builtin_amdgcn_wmma_f32_16x16x32_bf16(false, (a), false, (b), (short)0, (c), false, false)

// Async copy 16B global -> LDS (per-lane addresses), tracked by ASYNCcnt.
// CDNA5 ISA: LDS address comes from the VDST VGPR; GV mode addr.
// LDS byte offset = low 32 bits of the generic shared-aperture address.
__device__ __forceinline__ void async_b128_to_lds(void* lds_ptr, const void* gptr) {
  unsigned ldsa = (unsigned)(unsigned long long)lds_ptr;
  asm volatile("global_load_async_to_lds_b128 %0, %1, off"
               :: "v"(ldsa), "v"(gptr) : "memory");
}
__device__ __forceinline__ void wait_asynccnt0() {
#if __has_builtin(__builtin_amdgcn_s_wait_asynccnt)
  __builtin_amdgcn_s_wait_asynccnt(0);
#else
  asm volatile("s_wait_asynccnt 0" ::: "memory");
#endif
}

// ---------------------------------------------------------------------------
// Pre-pass 1: hidden fp32 -> bf16, 8 elements per thread (one-time, mem-bound)
// ---------------------------------------------------------------------------
__global__ __launch_bounds__(256) void cvt_hidden_kernel(
    const float* __restrict__ src, unsigned short* __restrict__ dst) {
  size_t idx = ((size_t)blockIdx.x * 256 + threadIdx.x) * 8;
  const float2* s = (const float2*)(src + idx);
  float2 f0 = s[0], f1 = s[1], f2 = s[2], f3 = s[3];
  uint4 o;
  o.x = packbf(f0.x, f0.y); o.y = packbf(f1.x, f1.y);
  o.z = packbf(f2.x, f2.y); o.w = packbf(f3.x, f3.y);
  *(uint4*)(dst + idx) = o;
}

// ---------------------------------------------------------------------------
// Pre-pass 2: W [h][k][e] fp32 -> wbt [h][e][k] bf16 (one-time transpose)
// Thread handles (h, e, 8 consecutive k); reads coalesced over e.
// ---------------------------------------------------------------------------
__global__ __launch_bounds__(256) void cvt_w_kernel(
    const float* __restrict__ Wsrc, unsigned short* __restrict__ wbt) {
  int idx  = blockIdx.x * 256 + threadIdx.x;      // 16*128*192 = 393216 total
  int e    = idx % E3_;
  int rest = idx / E3_;
  int k8   = (rest & 127) * 8;
  int h    = rest >> 7;
  float f[8];
#pragma unroll
  for (int j = 0; j < 8; ++j) f[j] = Wsrc[((size_t)h * HID_ + k8 + j) * E3_ + e];
  uint4 o;
  o.x = packbf(f[0], f[1]); o.y = packbf(f[2], f[3]);
  o.z = packbf(f[4], f[5]); o.w = packbf(f[6], f[7]);
  *(uint4*)&wbt[((size_t)h * E3_ + e) * HID_ + k8] = o;
}

// ---------------------------------------------------------------------------
// Kernel 1: qkv[b,h,n,e] = sum_D hidden[b,n,D] * W[h,D,e],  e in [0,192)
// Block = 256 threads (8 waves) = one (b,h) x 128 rows.
// Wave w: rows (w&3)*32..+32, cols (w>>2)*96..+96  (2x6 = 12 accum tiles).
// Double-buffered async-to-LDS staging: DMA of chunk k+1 overlaps wmma of k.
// ---------------------------------------------------------------------------
__global__ __launch_bounds__(256) void qkv_proj_kernel(
    const unsigned short* __restrict__ hbf, const unsigned short* __restrict__ wbt,
    unsigned short* __restrict__ Qs, unsigned short* __restrict__ Ks,
    unsigned short* __restrict__ Vt) {
  __shared__ __align__(16) unsigned short lh[2][128][40];   // hidden tiles, padded
  __shared__ __align__(16) unsigned short lw[2][192][40];   // W tiles [e][k], padded

  const int ntile = N_ / 128;                 // 16
  const int bh   = blockIdx.x / ntile;
  const int tile = blockIdx.x % ntile;
  const int b = bh / H_, h = bh % H_;
  const int row0 = tile * 128;
  const int tid = threadIdx.x;
  const int w = tid >> 5, lane = tid & 31, ln = lane & 15, g = lane >> 4;
  const int rw = w & 3;        // row group: 32 rows
  const int cw = w >> 2;       // col group: 96 cols

  const v8f vzero = {0.f,0.f,0.f,0.f,0.f,0.f,0.f,0.f};
  v8f acc[12];
#pragma unroll
  for (int i = 0; i < 12; ++i) acc[i] = vzero;

  const unsigned short* hp = hbf + (size_t)(b * N_ + row0) * HID_;
  const unsigned short* wp = wbt + (size_t)h * E3_ * HID_;

  auto stage = [&](int buf, int kk) {
#pragma unroll
    for (int ii = 0; ii < 2; ++ii) {          // 128 rows x 64B
      int idx = tid + ii * 256;
      int r = idx >> 2, k8 = (idx & 3) * 8;
      async_b128_to_lds(&lh[buf][r][k8], hp + (size_t)r * HID_ + kk + k8);
    }
#pragma unroll
    for (int ii = 0; ii < 3; ++ii) {          // 192 e-rows x 64B
      int idx = tid + ii * 256;
      int e = idx >> 2, k8 = (idx & 3) * 8;
      async_b128_to_lds(&lw[buf][e][k8], wp + (size_t)e * HID_ + kk + k8);
    }
  };

  stage(0, 0);
  for (int kk = 0; kk < HID_; kk += 32) {
    const int cur = (kk >> 5) & 1;
    wait_asynccnt0();            // this wave's copies for buf[cur] are done
    __syncthreads();             // => everyone's copies for buf[cur] are done
    if (kk + 32 < HID_) stage(cur ^ 1, kk + 32);   // DMA next chunk, overlapped

    // Load all fragments first, then burst 12 wmmas (reuse 2A x 6B).
    BF16Frag a[2], bm[6];
#pragma unroll
    for (int i = 0; i < 2; ++i) {
      const int ar = rw * 32 + i * 16 + ln;
      a[i].q[0] = *(const uint4*)&lh[cur][ar][g * 8];
      a[i].q[1] = *(const uint4*)&lh[cur][ar][16 + g * 8];
    }
#pragma unroll
    for (int j = 0; j < 6; ++j) {
      const int e = cw * 96 + j * 16 + ln;
      bm[j].q[0] = *(const uint4*)&lw[cur][e][g * 8];
      bm[j].q[1] = *(const uint4*)&lw[cur][e][16 + g * 8];
    }
#pragma unroll
    for (int i = 0; i < 2; ++i)
#pragma unroll
      for (int j = 0; j < 6; ++j)
        acc[i * 6 + j] = WMMA_BF16(a[i].v, bm[j].v, acc[i * 6 + j]);
  }

  // Epilogue: one wave-uniform branch on cw; routing resolved per unrolled j.
  // Q,K stored [b,h,n,d]; V stored transposed [b,h,d,n] for async flash staging.
  const size_t bhN = (size_t)bh * N_;
  if (cw == 0) {                 // e in [0,96): j<4 -> Q, j=4,5 -> K
#pragma unroll
    for (int i = 0; i < 2; ++i)
#pragma unroll
      for (int j = 0; j < 6; ++j) {
        unsigned short* dst = (j < 4) ? Qs : Ks;
        const int d = (j < 4) ? (j * 16 + ln) : ((j - 4) * 16 + ln);
#pragma unroll
        for (int r = 0; r < 8; ++r) {
          int rowm = row0 + rw * 32 + i * 16 + r + 8 * g;
          dst[(bhN + rowm) * D_ + d] = (unsigned short)f2bf1(acc[i * 6 + j][r]);
        }
      }
  } else {                       // e in [96,192): j<2 -> K, j>=2 -> V (transposed)
#pragma unroll
    for (int i = 0; i < 2; ++i)
#pragma unroll
      for (int j = 0; j < 6; ++j) {
#pragma unroll
        for (int r = 0; r < 8; ++r) {
          int rowm = row0 + rw * 32 + i * 16 + r + 8 * g;
          unsigned short val = (unsigned short)f2bf1(acc[i * 6 + j][r]);
          if (j < 2) {
            const int d = 32 + j * 16 + ln;
            Ks[(bhN + rowm) * D_ + d] = val;
          } else {
            const int d = (j - 2) * 16 + ln;
            Vt[((size_t)bh * D_ + d) * N_ + rowm] = val;
          }
        }
      }
  }
}

// ---------------------------------------------------------------------------
// Kernel 2: flash attention. Block = (b,h, 128 query rows), 8 waves x 16 rows.
// 64-key blocks: S = Q Kt (8 wmma), online softmax, O += P V (8 wmma).
// K and Vt both staged with double-buffered async-to-LDS (no transpose work).
// ---------------------------------------------------------------------------
__global__ __launch_bounds__(256) void flash_attn_kernel(
    const unsigned short* __restrict__ Qs, const unsigned short* __restrict__ Ks,
    const unsigned short* __restrict__ Vt, const int* __restrict__ amask,
    float* __restrict__ out) {
  __shared__ __align__(16) unsigned short lk[2][64][72];    // K block [key][d]
  __shared__ __align__(16) unsigned short lvt[2][64][72];   // Vt block [d][key]
  __shared__ __align__(16) unsigned short lp[8][16][72];    // per-wave P staging

  const int ntile = N_ / 128;
  const int bh   = blockIdx.x / ntile;
  const int tile = blockIdx.x % ntile;
  const int b = bh / H_, h = bh % H_;
  const int q0 = tile * 128;
  const int tid = threadIdx.x;
  const int w = tid >> 5, lane = tid & 31, ln = lane & 15, g = lane >> 4;
  const size_t bhN = (size_t)bh * N_;

  const v8f vzero = {0.f,0.f,0.f,0.f,0.f,0.f,0.f,0.f};
  v8f o[4];
#pragma unroll
  for (int t = 0; t < 4; ++t) o[t] = vzero;
  float m[8], l[8];
#pragma unroll
  for (int r = 0; r < 8; ++r) { m[r] = -1e30f; l[r] = 0.f; }

  // Q fragments for both 32-wide d-chunks, resident for the whole loop
  BF16Frag qa[2];
  const unsigned short* qp = Qs + (bhN + q0 + w * 16 + ln) * D_;
#pragma unroll
  for (int c = 0; c < 2; ++c) {
    qa[c].q[0] = *(const uint4*)(qp + c * 32 + g * 8);
    qa[c].q[1] = *(const uint4*)(qp + c * 32 + 16 + g * 8);
  }

  auto stageKV = [&](int buf, int jb) {
#pragma unroll
    for (int ii = 0; ii < 2; ++ii) {          // K: 64 keys x 128B
      int idx = tid + ii * 256;
      int i = idx >> 3, d8 = (idx & 7) * 8;
      async_b128_to_lds(&lk[buf][i][d8], &Ks[(bhN + jb + i) * D_ + d8]);
    }
#pragma unroll
    for (int ii = 0; ii < 2; ++ii) {          // Vt: 64 d-rows x 128B
      int idx = tid + ii * 256;
      int d = idx >> 3, k8 = (idx & 7) * 8;
      async_b128_to_lds(&lvt[buf][d][k8], &Vt[((size_t)bh * D_ + d) * N_ + jb + k8]);
    }
  };

  stageKV(0, 0);
  for (int jb = 0; jb < N_; jb += 64) {
    const int cur = (jb >> 6) & 1;
    wait_asynccnt0();
    __syncthreads();
    if (jb + 64 < N_) stageKV(cur ^ 1, jb + 64);   // DMA next block, overlapped

    int mk[4];
#pragma unroll
    for (int t = 0; t < 4; ++t) mk[t] = amask[b * N_ + jb + t * 16 + ln];

    // S = Q Kt : four 16-key tiles, K-dim 64 split into two 32-chunks
    v8f s[4];
#pragma unroll
    for (int t = 0; t < 4; ++t) s[t] = vzero;
#pragma unroll
    for (int c = 0; c < 2; ++c) {
#pragma unroll
      for (int t = 0; t < 4; ++t) {
        BF16Frag kb;
        kb.q[0] = *(const uint4*)&lk[cur][t * 16 + ln][c * 32 + g * 8];
        kb.q[1] = *(const uint4*)&lk[cur][t * 16 + ln][c * 32 + 16 + g * 8];
        s[t] = WMMA_BF16(qa[c].v, kb.v, s[t]);
      }
    }

    // Online softmax in C-fragment layout: lane holds rows r+8g at column ln.
#pragma unroll
    for (int r = 0; r < 8; ++r) {
      float av[4];
#pragma unroll
      for (int t = 0; t < 4; ++t) {
        av[t] = s[t][r] * SCALE_;
        if (mk[t] == 0) av[t] = -1e30f;
      }
      float bmax = fmaxf(fmaxf(av[0], av[1]), fmaxf(av[2], av[3]));
#pragma unroll
      for (int off = 1; off < 16; off <<= 1)
        bmax = fmaxf(bmax, __shfl_xor(bmax, off, 32));
      float mn    = fmaxf(m[r], bmax);
      float alpha = __expf(m[r] - mn);
      float ev[4];
      float rs = 0.f;
#pragma unroll
      for (int t = 0; t < 4; ++t) { ev[t] = __expf(av[t] - mn); rs += ev[t]; }
#pragma unroll
      for (int off = 1; off < 16; off <<= 1)
        rs += __shfl_xor(rs, off, 32);
      l[r] = l[r] * alpha + rs;
      m[r] = mn;
#pragma unroll
      for (int t = 0; t < 4; ++t) o[t][r] *= alpha;
#pragma unroll
      for (int t = 0; t < 4; ++t)
        lp[w][r + 8 * g][t * 16 + ln] = (unsigned short)f2bf1(ev[t]);
    }
    __syncthreads();   // order P stores vs. A-fragment reloads

    // O += P (16x64) x V (64x64): two 32-key A chunks x four d tiles
#pragma unroll
    for (int kc = 0; kc < 2; ++kc) {
      BF16Frag pa;
      pa.q[0] = *(const uint4*)&lp[w][ln][kc * 32 + g * 8];
      pa.q[1] = *(const uint4*)&lp[w][ln][kc * 32 + 16 + g * 8];
#pragma unroll
      for (int t = 0; t < 4; ++t) {
        BF16Frag vb;
        vb.q[0] = *(const uint4*)&lvt[cur][t * 16 + ln][kc * 32 + g * 8];
        vb.q[1] = *(const uint4*)&lvt[cur][t * 16 + ln][kc * 32 + 16 + g * 8];
        o[t] = WMMA_BF16(pa.v, vb.v, o[t]);
      }
    }
  }

  // Epilogue: out[b, n, h*64 + d] = O / l
#pragma unroll
  for (int r = 0; r < 8; ++r) {
    float inv = (l[r] > 0.f) ? 1.f / l[r] : 0.f;
    int rowm = q0 + w * 16 + r + 8 * g;
    float* op = out + (size_t)(b * N_ + rowm) * HID_ + h * D_;
#pragma unroll
    for (int t = 0; t < 4; ++t)
      op[t * 16 + ln] = o[t][r] * inv;
  }
}

// ---------------------------------------------------------------------------
extern "C" void kernel_launch(void* const* d_in, const int* in_sizes, int n_in,
                              void* d_out, int out_size, void* d_ws, size_t ws_size,
                              hipStream_t stream) {
  const float* hs   = (const float*)d_in[0];   // [B, N, 1024]
  const int*   mask = (const int*)d_in[1];     // [B, N]
  const float* W    = (const float*)d_in[2];   // [H, 1024, 192]
  float*       out  = (float*)d_out;           // [B, N, 1024]

  unsigned short* Qs  = (unsigned short*)d_ws;        // PER_ each
  unsigned short* Ks  = Qs + PER_;
  unsigned short* Vt  = Ks + PER_;                    // [b,h,d,n]
  unsigned short* hbf = Vt + PER_;                    // B*N*HID = PER_ elems
  unsigned short* wbt = hbf + PER_;                   // H*E3*HID = 3145728 elems
                                                      // total ~38 MB of ws

  cvt_hidden_kernel<<<(int)(PER_ / 8 / 256), 256, 0, stream>>>(hs, hbf);
  cvt_w_kernel<<<(H_ * 128 * E3_) / 256, 256, 0, stream>>>(W, wbt);

  const int blocks = B_ * H_ * (N_ / 128);            // 512
  qkv_proj_kernel<<<blocks, 256, 0, stream>>>(hbf, wbt, Qs, Ks, Vt);
  flash_attn_kernel<<<blocks, 256, 0, stream>>>(Qs, Ks, Vt, mask, out);
}
